// EdgeConvGNNClassifier_47493748359664
// MI455X (gfx1250) — matrix operand order repaired and checked by reading, not compiled, not run on hardware
//
#include <hip/hip_runtime.h>
#include <math.h>

typedef __attribute__((ext_vector_type(16))) _Float16 v16h;
typedef __attribute__((ext_vector_type(8)))  float    v8f;

// ---- monotone float <-> uint key map so unsigned atomic max == float max ----
__device__ __forceinline__ unsigned fkey(float f) {
    unsigned b = __float_as_uint(f);
    return b ^ ((unsigned)((int)b >> 31) | 0x80000000u);
}
__device__ __forceinline__ float funkey(unsigned k) {
    unsigned b = (k & 0x80000000u) ? (k ^ 0x80000000u) : ~k;
    return __uint_as_float(b);
}

// Weight slot layout in workspace (per matrix): 32 lanes x 16 f16 (1024 B, WMMA
// B layout: b[j] = W[8h+j][n], b[8+j] = W[16+8h+j][n], zero-padded) followed by 32 per-lane bias
// floats (128 B).  Slot stride 1152 B (32B aligned).
#define WSLOT 1152

__global__ void prep_weights(const float* __restrict__ W1a, const float* __restrict__ b1a,
                             const float* __restrict__ W1b, const float* __restrict__ b1b,
                             const float* __restrict__ W2a, const float* __restrict__ b2a,
                             const float* __restrict__ W2b, const float* __restrict__ b2b,
                             const float* __restrict__ W3,  const float* __restrict__ b3,
                             unsigned char* __restrict__ wbuf)
{
    const int tid  = threadIdx.x;
    const int slot = tid >> 5;
    const int lane = tid & 31;
    if (slot >= 5) return;

    const float* W; const float* bias; int K; int NOUT;
    switch (slot) {
        case 0: W = W1a; bias = b1a; K = 8;  NOUT = 16; break;
        case 1: W = W1b; bias = b1b; K = 16; NOUT = 16; break;
        case 2: W = W2a; bias = b2a; K = 32; NOUT = 16; break;
        case 3: W = W2b; bias = b2b; K = 16; NOUT = 16; break;
        default:W = W3;  bias = b3;  K = 32; NOUT = 1;  break;
    }
    const int h = lane >> 4, n = lane & 15;
    v16h b;
    for (int j = 0; j < 16; ++j) {
        const int k = (j < 8) ? (8 * h + j) : (16 + 8 * h + (j - 8));
        float w = 0.f;
        if (k < K && n < NOUT) w = W[k * NOUT + n];
        b[j] = (_Float16)w;
    }
    *(v16h*)(wbuf + slot * WSLOT + lane * 32) = b;
    *(float*)(wbuf + slot * WSLOT + 1024 + lane * 4) = (n < NOUT) ? bias[n] : 0.f;
}

// One wave handles a tile of 16 edges: M = edge-in-tile, N = output feature.
// A (16xK f16): lane l holds row M=l%16, h=l/16; a[j] -> K=8h+j, a[8+j] -> K=16+8h+j.
// C/D (16x16 f32): lane l, vgpr r -> element [M=r+8h][N=l%16].
template<int C_IN, bool TWO_MLP, bool SIG>
__global__ void __launch_bounds__(256)
edge_conv_wmma(const float* __restrict__ xf,
               const int*   __restrict__ src, const int* __restrict__ dst,
               const unsigned char* __restrict__ slotA,
               const unsigned char* __restrict__ slotB,
               unsigned*    __restrict__ agg,
               int n_edges, int tiles_per_wave, int agg_stride)
{
    __shared__ float lds_h[8][16 * 17];   // per-wave 16x16 hidden tile (padded)
    __shared__ int   lds_d[8][16];        // per-wave dst node ids

    const int lane = threadIdx.x & 31;
    const int wv   = threadIdx.x >> 5;
    const int h    = lane >> 4;           // half of wave
    const int n    = lane & 15;           // N column / row-in-tile for gathers
    const int gw   = blockIdx.x * (blockDim.x >> 5) + wv;

    // ---- prepacked WMMA B operands + per-lane biases: 2x b128 + b32 each ----
    const v16h  bA    = *(const v16h*)(slotA + lane * 32);
    const float biasA = *(const float*)(slotA + 1024 + lane * 4);
    v16h  bB;
    float biasB = 0.f;
    if (TWO_MLP) {
        bB    = *(const v16h*)(slotB + lane * 32);
        biasB = *(const float*)(slotB + 1024 + lane * 4);
    }

    for (int t = 0; t < tiles_per_wave; ++t) {
        const int e0 = (gw * tiles_per_wave + t) * 16;
        const int eM = e0 + n;                 // this lane's gather edge (row M=n)
        const bool ev = (eM < n_edges);

        int s_i = 0, d_i = 0;
        if (ev && (C_IN == 16 || h == 0)) { s_i = src[eM]; d_i = dst[eM]; }
        if (h == 0) lds_d[wv][n] = ev ? d_i : -1;

        // ---- build A = [xi, xj - xi] in f16 via b128 gathers ----
        v16h a;
        #pragma unroll
        for (int j = 0; j < 16; ++j) a[j] = (_Float16)0.f;
        if (C_IN == 4) {
            if (ev && h == 0) {                       // K=0..7 live, rest zero
                const float4 xi = *(const float4*)(xf + (size_t)d_i * 4);
                const float4 xj = *(const float4*)(xf + (size_t)s_i * 4);
                a[0] = (_Float16)xi.x; a[1] = (_Float16)xi.y;
                a[2] = (_Float16)xi.z; a[3] = (_Float16)xi.w;
                a[4] = (_Float16)(xj.x - xi.x); a[5] = (_Float16)(xj.y - xi.y);
                a[6] = (_Float16)(xj.z - xi.z); a[7] = (_Float16)(xj.w - xi.w);
            }
        } else if (ev) {                              // C_IN==16, K=0..31 live
            const float4* pi = (const float4*)(xf + (size_t)d_i * 16 + 8 * h);
            const float4* pj = (const float4*)(xf + (size_t)s_i * 16 + 8 * h);
            const float4 xi0 = pi[0], xi1 = pi[1];
            const float4 xj0 = pj[0], xj1 = pj[1];
            a[0]  = (_Float16)xi0.x; a[1]  = (_Float16)xi0.y;
            a[2]  = (_Float16)xi0.z; a[3]  = (_Float16)xi0.w;
            a[4]  = (_Float16)xi1.x; a[5]  = (_Float16)xi1.y;
            a[6]  = (_Float16)xi1.z; a[7]  = (_Float16)xi1.w;
            a[8]  = (_Float16)(xj0.x - xi0.x); a[9]  = (_Float16)(xj0.y - xi0.y);
            a[10] = (_Float16)(xj0.z - xi0.z); a[11] = (_Float16)(xj0.w - xi0.w);
            a[12] = (_Float16)(xj1.x - xi1.x); a[13] = (_Float16)(xj1.y - xi1.y);
            a[14] = (_Float16)(xj1.z - xi1.z); a[15] = (_Float16)(xj1.w - xi1.w);
        }

        // ---- GEMM 1: msg @ Wa + ba ----
        v8f c;
        #pragma unroll
        for (int r = 0; r < 8; ++r) c[r] = biasA;
        c = __builtin_amdgcn_wmma_f32_16x16x32_f16(false, a, false, bA,
                                                   (short)0, c, false, false);

        if (TWO_MLP) {
            // relu, re-layout D -> A via per-wave padded LDS tile
            #pragma unroll
            for (int r = 0; r < 8; ++r) {
                const float v = c[r] > 0.f ? c[r] : 0.f;
                lds_h[wv][(r + 8 * h) * 17 + n] = v;
            }
            asm volatile("s_wait_dscnt 0" ::: "memory");
            v16h a2;
            #pragma unroll
            for (int j = 0; j < 16; ++j) a2[j] = (_Float16)0.f;
            #pragma unroll
            for (int j = 0; j < 8; ++j)                 // K = 8h+j < 16
                a2[j] = (_Float16)lds_h[wv][n * 17 + 8 * h + j];

            // ---- GEMM 2: hidden @ Wb + bb ----
            v8f c2;
            #pragma unroll
            for (int r = 0; r < 8; ++r) c2[r] = biasB;
            c2 = __builtin_amdgcn_wmma_f32_16x16x32_f16(false, a2, false, bB,
                                                        (short)0, c2, false, false);
            c = c2;
        }

        if (SIG) {
            #pragma unroll
            for (int r = 0; r < 8; ++r) c[r] = 1.f / (1.f + __expf(-c[r]));
        }

        // ---- max-scatter: one global_atomic_max_u32 per output element ----
        asm volatile("s_wait_dscnt 0" ::: "memory");
        if (!SIG || n == 0) {
            #pragma unroll
            for (int r = 0; r < 8; ++r) {
                const int M  = r + 8 * h;
                const int dn = lds_d[wv][M];
                if (dn >= 0)
                    atomicMax(&agg[(size_t)dn * agg_stride + (SIG ? 0 : n)],
                              fkey(c[r]));
            }
        }
    }
}

__global__ void init_agg(unsigned* __restrict__ agg, int n) {
    const int i = blockIdx.x * blockDim.x + threadIdx.x;
    if (i < n) agg[i] = 0u;
}

// x = relu(decode(max)), isolated nodes (sentinel 0) -> 0; re-zero agg for next layer
__global__ void fixup_relu(unsigned* __restrict__ agg, float* __restrict__ X, int n) {
    const int i = blockIdx.x * blockDim.x + threadIdx.x;
    if (i < n) {
        const unsigned k = agg[i];
        const float v = (k == 0u) ? 0.f : funkey(k);
        X[i] = v > 0.f ? v : 0.f;     // relu also maps -inf -> 0
        agg[i] = 0u;
    }
}

__global__ void fixup_out(const unsigned* __restrict__ agg, float* __restrict__ out, int n) {
    const int i = blockIdx.x * blockDim.x + threadIdx.x;
    if (i < n) {
        const unsigned k = agg[i];
        out[i] = (k == 0u) ? 0.f : funkey(k);   // sigmoid outputs are finite & > 0
    }
}

extern "C" void kernel_launch(void* const* d_in, const int* in_sizes, int n_in,
                              void* d_out, int out_size, void* d_ws, size_t ws_size,
                              hipStream_t stream) {
    const float* x   = (const float*)d_in[0];
    const int*   ei  = (const int*)d_in[1];   // [2, E] int32
    const float* W1a = (const float*)d_in[3];
    const float* b1a = (const float*)d_in[4];
    const float* W1b = (const float*)d_in[5];
    const float* b1b = (const float*)d_in[6];
    const float* W2a = (const float*)d_in[7];
    const float* b2a = (const float*)d_in[8];
    const float* W2b = (const float*)d_in[9];
    const float* b2b = (const float*)d_in[10];
    const float* W3  = (const float*)d_in[11];
    const float* b3  = (const float*)d_in[12];

    const int N = in_sizes[0] / 4;
    const int E = in_sizes[1] / 2;
    const int* src = ei;
    const int* dst = ei + E;

    unsigned*      agg  = (unsigned*)d_ws;                                   // N*16 keys
    float*         X    = (float*)((char*)d_ws + (size_t)N * 16 * 4);        // N*16 feats
    unsigned char* wbuf = (unsigned char*)d_ws + (size_t)N * 16 * 8;         // 5*1152 B
    float*         out  = (float*)d_out;

    const int nAgg = N * 16;
    const int TPW  = 4;                               // 64 edges per wave
    const int waves  = (E + TPW * 16 - 1) / (TPW * 16);
    const int blocks = (waves + 7) / 8;               // 8 waves (256 thr) per block
    const int eb = 256;

    prep_weights<<<1, 256, 0, stream>>>(W1a, b1a, W1b, b1b, W2a, b2a,
                                        W2b, b2b, W3, b3, wbuf);
    init_agg<<<(nAgg + eb - 1) / eb, eb, 0, stream>>>(agg, nAgg);

    edge_conv_wmma<4, true, false><<<blocks, 256, 0, stream>>>(
        x, src, dst, wbuf + 0 * WSLOT, wbuf + 1 * WSLOT, agg, E, TPW, 16);
    fixup_relu<<<(nAgg + eb - 1) / eb, eb, 0, stream>>>(agg, X, nAgg);

    edge_conv_wmma<16, true, false><<<blocks, 256, 0, stream>>>(
        X, src, dst, wbuf + 2 * WSLOT, wbuf + 3 * WSLOT, agg, E, TPW, 16);
    fixup_relu<<<(nAgg + eb - 1) / eb, eb, 0, stream>>>(agg, X, nAgg);

    edge_conv_wmma<16, false, true><<<blocks, 256, 0, stream>>>(
        X, src, dst, wbuf + 4 * WSLOT, wbuf + 4 * WSLOT, agg, E, TPW, 1);
    fixup_out<<<(N + eb - 1) / eb, eb, 0, stream>>>(agg, out, N);
}
